// PRNN_19928648253954
// MI455X (gfx1250) — matrix operand order repaired and checked by reading, not compiled, not run on hardware
//
#include <hip/hip_runtime.h>
#include <hip/hip_bf16.h>
#include <math.h>

// ---------------------------------------------------------------------------
// PRNN on MI455X (gfx1250): bf16 WMMA + LDS A-panel staging.
//   enc = x @ W_enc^T + b_enc            (one big WMMA GEMM, T*B rows)
//   scan: h = (1-s)h + s*a*tanh(e_t + h @ W_rec^T + b_rec)   (512 step kernels)
//   y   = hist @ W_dec^T + b_dec         (one big WMMA GEMM)
// Blocks own one 16-row m-tile; its A panel is staged in LDS (async copy for
// bf16 sources), and each wave accumulates two 16x16 n-tiles off one A frag.
// ---------------------------------------------------------------------------

typedef __attribute__((ext_vector_type(16))) __bf16 v16bf;
typedef __attribute__((ext_vector_type(8)))  __bf16 v8bf;
typedef __attribute__((ext_vector_type(8)))  float  v8f;
typedef __attribute__((ext_vector_type(4)))  float  v4f;

static constexpr int Bsz  = 256;
static constexpr int Tsz  = 512;
static constexpr int NIsz = 512;
static constexpr int NHsz = 2048;
static constexpr int NOsz = 256;
static constexpr long TBsz = (long)Tsz * Bsz;   // 131072 rows

// LDS row strides (elements), padded so the 16 A-rows hit distinct banks.
static constexpr int LDH = NHsz + 8;            // 2056 (row = 4112 B)
static constexpr int LDI = NIsz + 8;            // 520  (row = 1040 B)
static constexpr size_t SMEM_H = (size_t)16 * LDH * 2;  // 65792 B
static constexpr size_t SMEM_I = (size_t)16 * LDI * 2;  // 16640 B

// ---------------- wave-level WMMA helpers (A and B use contiguous k order) ----------

__device__ __forceinline__ v8f wmma_bf16f32(v16bf a, v16bf b, v8f c) {
  return __builtin_amdgcn_wmma_f32_16x16x32_bf16(false, a, false, b, (short)0, c,
                                                 false, false);
}

// A fragment 16x32 bf16: lane groups 0/1 take K chunks {0-15}/{16-31}.
// Works for global or LDS-derived pointers (addrspace inferred -> ds_load_b128).
__device__ __forceinline__ v16bf frag_a_bf16(const __bf16* row, int kb, int grp) {
  const __bf16* p = row + kb + grp * 16;
  v8bf lo = *(const v8bf*)p;
  v8bf hi = *(const v8bf*)(p + 8);
  v16bf r;
#pragma unroll
  for (int i = 0; i < 8; ++i) { r[i] = lo[i]; r[i + 8] = hi[i]; }
  return r;
}

// Same, but source row is f32 (convert in flight -> bf16).
__device__ __forceinline__ v16bf frag_a_f32(const float* row, int kb, int grp) {
  const float* p = row + kb + grp * 16;
  v8f lo = *(const v8f*)p;
  v8f hi = *(const v8f*)(p + 8);
  v16bf r;
#pragma unroll
  for (int i = 0; i < 8; ++i) { r[i] = (__bf16)lo[i]; r[i + 8] = (__bf16)hi[i]; }
  return r;
}

// B fragment 32x16 bf16: lane holds column n = lane&15, K contiguous.
__device__ __forceinline__ v16bf frag_b_bf16(const __bf16* wcolbase, int kb) {
  return *(const v16bf*)(wcolbase + kb);
}

// Async bulk copy: 16 bf16 rows of `width` elems, global -> LDS (row stride
// ldsStride elems). 256 threads; tracked on ASYNCcnt; barrier follows.
__device__ __forceinline__ void stage_panel_async(const __bf16* __restrict__ gsrc,
                                                  size_t gRowStride,
                                                  __bf16* lds, int ldsStride,
                                                  int width /* multiple of 128 */) {
  const int tid = threadIdx.x;
  const int row = tid >> 4;          // 0..15
  const int seg = tid & 15;          // 0..15
  const int segW = width / 16;       // elems per segment (mult of 8)
  const __bf16* src = gsrc + (size_t)row * gRowStride + seg * segW;
  unsigned dst = (unsigned)(uintptr_t)(lds + row * ldsStride + seg * segW);
#pragma unroll
  for (int i = 0; i < 16; ++i) {     // 16 x b128 per thread = 256 B
    asm volatile("global_load_async_to_lds_b128 %0, %1, off"
                 :: "v"(dst + i * 16),
                    "v"((unsigned long long)(const void*)(src + i * 8))
                 : "memory");
  }
  asm volatile("s_wait_asynccnt 0x0" ::: "memory");
}

// ---------------- utility kernels ------------------------------------------

__global__ void prnn_cvt_f32_to_bf16(const float* __restrict__ src,
                                     __bf16* __restrict__ dst, int n) {
  int i = blockIdx.x * blockDim.x + threadIdx.x;
  int stride = gridDim.x * blockDim.x;
  for (; i < n; i += stride) dst[i] = (__bf16)src[i];
}

__global__ void prnn_zero_f32(float* __restrict__ p, int n) {
  int i = blockIdx.x * blockDim.x + threadIdx.x;
  int stride = gridDim.x * blockDim.x;
  for (; i < n; i += stride) p[i] = 0.0f;
}

__global__ void prnn_zero_bf16(__bf16* __restrict__ p, int n) {
  int i = blockIdx.x * blockDim.x + threadIdx.x;
  int stride = gridDim.x * blockDim.x;
  for (; i < n; i += stride) p[i] = (__bf16)0.0f;
}

// ---------------- encoder: enc[t*B+b, :] = x[b,t,:] @ Wenc^T + b_enc --------
// Block: one m-tile (16 rows of x, K=512) + 16 n-tiles (8 waves x 2).
// Stage x-panel converted to bf16 in LDS, then K-loop reads ds_load_b128.

__global__ __launch_bounds__(256) void prnn_encoder_kernel(
    const float* __restrict__ x,        // [B, T, NI] f32
    const __bf16* __restrict__ Wenc,    // [NH, NI] bf16
    const float* __restrict__ b_enc,    // [NH]
    __bf16* __restrict__ enc) {         // [T*B, NH] bf16 (time-major rows)
  extern __shared__ __bf16 lx[];        // 16 x LDI
  const int tid    = threadIdx.x;
  const long mTile = blockIdx.x >> 3;   // 0..8191
  const int nGroup = blockIdx.x & 7;    // 0..7
  const long mBase = mTile * 16;

  // stage + convert the x panel: 16 rows x 512 f32 -> bf16
  {
    const int row = tid >> 4, seg = tid & 15;     // 32 f32 per thread
    const long m = mBase + row;
    const int t = (int)(m / Bsz), b = (int)(m % Bsz);
    const float* src = x + ((size_t)b * Tsz + t) * NIsz + seg * 32;
    __bf16* dst = lx + row * LDI + seg * 32;
#pragma unroll
    for (int i = 0; i < 8; ++i) {
      v4f v = *(const v4f*)(src + i * 4);
#pragma unroll
      for (int j = 0; j < 4; ++j) dst[i * 4 + j] = (__bf16)v[j];
    }
  }
  __syncthreads();

  const int lane = tid & 31, col = lane & 15, grp = lane >> 4;
  const int w = tid >> 5;                         // wave 0..7
  const int nBase = nGroup * 256 + w * 32;        // two tiles: nBase, nBase+16

  const __bf16* lrow  = lx + (size_t)col * LDI;
  const __bf16* bcol0 = Wenc + (size_t)(nBase + col) * NIsz + grp * 16;
  const __bf16* bcol1 = bcol0 + (size_t)16 * NIsz;

  v8f acc0, acc1;
  const float bias0 = b_enc[nBase + col], bias1 = b_enc[nBase + 16 + col];
#pragma unroll
  for (int r = 0; r < 8; ++r) { acc0[r] = bias0; acc1[r] = bias1; }

#pragma unroll 4
  for (int kb = 0; kb < NIsz; kb += 32) {
    v16bf af = frag_a_bf16(lrow, kb, grp);        // ds_load_b128 x2
    acc0 = wmma_bf16f32(af, frag_b_bf16(bcol0, kb), acc0);
    acc1 = wmma_bf16f32(af, frag_b_bf16(bcol1, kb), acc1);
  }

  __bf16* orow = enc + (size_t)(mBase + grp * 8) * NHsz + nBase + col;
#pragma unroll
  for (int r = 0; r < 8; ++r) {
    orow[(size_t)r * NHsz]      = (__bf16)acc0[r];
    orow[(size_t)r * NHsz + 16] = (__bf16)acc1[r];
  }
}

// ---------------- recurrent step (fast path) -------------------------------
// pre = e_t + h_in @ Wrec^T + b_rec ; h = (1-s)h + s*a*tanh(pre)
// Block: m-tile (16 of 256 rows) x 256-wide n-group; h panel async-staged.

__global__ __launch_bounds__(256) void prnn_step_kernel(
    __bf16* __restrict__ hist,          // [T*B, NH] bf16 (slot t: e in, h out)
    const __bf16* __restrict__ h_in,    // [B, NH] bf16
    __bf16* __restrict__ h_out,         // [B, NH] bf16
    float* __restrict__ h_f32,          // [B, NH] f32
    const __bf16* __restrict__ Wrec,    // [NH, NH] bf16
    const float* __restrict__ b_rec,
    const float* __restrict__ dt,
    const float* __restrict__ a_gain,
    int t) {
  extern __shared__ __bf16 lh[];        // 16 x LDH
  const int tid    = threadIdx.x;
  const int mTile  = blockIdx.x >> 3;   // 0..15
  const int nGroup = blockIdx.x & 7;    // 0..7
  const int mBase  = mTile * 16;

  stage_panel_async(h_in + (size_t)mBase * NHsz, NHsz, lh, LDH, NHsz);
  __syncthreads();

  const int lane = tid & 31, col = lane & 15, grp = lane >> 4;
  const int w = tid >> 5;
  const int nBase = nGroup * 256 + w * 32;        // tiles nBase, nBase+16

  const float s  = 1.0f / (1.0f + expf(-dt[0]));
  const float av = a_gain[0];

  __bf16* eslot = hist + (size_t)t * Bsz * NHsz;

  v8f acc0, acc1;
  {
    const float bias0 = b_rec[nBase + col], bias1 = b_rec[nBase + 16 + col];
    const __bf16* crow = eslot + (size_t)(mBase + grp * 8) * NHsz + nBase + col;
#pragma unroll
    for (int r = 0; r < 8; ++r) {
      acc0[r] = (float)crow[(size_t)r * NHsz]      + bias0;
      acc1[r] = (float)crow[(size_t)r * NHsz + 16] + bias1;
    }
  }

  const __bf16* lrow  = lh + (size_t)col * LDH;
  const __bf16* bcol0 = Wrec + (size_t)(nBase + col) * NHsz + grp * 16;
  const __bf16* bcol1 = bcol0 + (size_t)16 * NHsz;

#pragma unroll 4
  for (int kb = 0; kb < NHsz; kb += 32) {
    v16bf af = frag_a_bf16(lrow, kb, grp);        // from LDS
    acc0 = wmma_bf16f32(af, frag_b_bf16(bcol0, kb), acc0);
    acc1 = wmma_bf16f32(af, frag_b_bf16(bcol1, kb), acc1);
  }

  const size_t base = (size_t)(mBase + grp * 8) * NHsz + nBase + col;
#pragma unroll
  for (int r = 0; r < 8; ++r) {
#pragma unroll
    for (int q = 0; q < 2; ++q) {
      const size_t idx = base + (size_t)r * NHsz + q * 16;
      const float pre  = (q == 0) ? acc0[r] : acc1[r];
      const float hold = h_f32[idx];
      const float hn   = (1.0f - s) * hold + s * av * tanhf(pre);
      h_f32[idx] = hn;
      const __bf16 hb = (__bf16)hn;
      h_out[idx] = hb;
      eslot[idx] = hb;            // history for batched decoder
    }
  }
}

// ---------------- batched decoder: y = hist @ Wdec^T + b_dec ---------------
// Block: one m-tile, all 16 n-tiles of NO (8 waves x 2); A panel async-staged.

__global__ __launch_bounds__(256) void prnn_decoder_kernel(
    const __bf16* __restrict__ hist,    // [T*B, NH] bf16
    const __bf16* __restrict__ Wdec,    // [NO, NH] bf16
    const float* __restrict__ b_dec,
    float* __restrict__ y) {            // [T*B, NO] f32
  extern __shared__ __bf16 la[];        // 16 x LDH
  const int tid = threadIdx.x;
  const long mBase = (long)blockIdx.x * 16;

  stage_panel_async(hist + (size_t)mBase * NHsz, NHsz, la, LDH, NHsz);
  __syncthreads();

  const int lane = tid & 31, col = lane & 15, grp = lane >> 4;
  const int w = tid >> 5;
  const int nBase = w * 32;

  const __bf16* lrow  = la + (size_t)col * LDH;
  const __bf16* bcol0 = Wdec + (size_t)(nBase + col) * NHsz + grp * 16;
  const __bf16* bcol1 = bcol0 + (size_t)16 * NHsz;

  v8f acc0, acc1;
  const float bias0 = b_dec[nBase + col], bias1 = b_dec[nBase + 16 + col];
#pragma unroll
  for (int r = 0; r < 8; ++r) { acc0[r] = bias0; acc1[r] = bias1; }

#pragma unroll 4
  for (int kb = 0; kb < NHsz; kb += 32) {
    v16bf af = frag_a_bf16(lrow, kb, grp);
    acc0 = wmma_bf16f32(af, frag_b_bf16(bcol0, kb), acc0);
    acc1 = wmma_bf16f32(af, frag_b_bf16(bcol1, kb), acc1);
  }

  float* orow = y + (size_t)(mBase + grp * 8) * NOsz + nBase + col;
#pragma unroll
  for (int r = 0; r < 8; ++r) {
    orow[(size_t)r * NOsz]      = acc0[r];
    orow[(size_t)r * NOsz + 16] = acc1[r];
  }
}

// ---------------- fused fallback (small workspace) -------------------------

__global__ __launch_bounds__(256) void prnn_step_fused_kernel(
    const float* __restrict__ x,
    const __bf16* __restrict__ Wenc,
    const __bf16* __restrict__ h_in,
    __bf16* __restrict__ h_out,
    float* __restrict__ h_f32,
    const __bf16* __restrict__ Wrec,
    const float* __restrict__ b_enc,
    const float* __restrict__ b_rec,
    const float* __restrict__ dt,
    const float* __restrict__ a_gain,
    int t) {
  const int lane = threadIdx.x & 31;
  const int col  = lane & 15;
  const int grp  = lane >> 4;
  const int gw   = blockIdx.x * (blockDim.x >> 5) + (threadIdx.x >> 5);
  const int nTiles = NHsz / 16;
  const int mBase = (gw / nTiles) * 16;
  const int nBase = (gw % nTiles) * 16;

  const float s  = 1.0f / (1.0f + expf(-dt[0]));
  const float av = a_gain[0];

  v8f acc;
  {
    const float bias = b_enc[nBase + col] + b_rec[nBase + col];
#pragma unroll
    for (int r = 0; r < 8; ++r) acc[r] = bias;
  }
  {
    const float* arow = x + ((size_t)(mBase + col) * Tsz + t) * NIsz;
    const __bf16* bcol = Wenc + (size_t)(nBase + col) * NIsz + grp * 16;
#pragma unroll 4
    for (int kb = 0; kb < NIsz; kb += 32)
      acc = wmma_bf16f32(frag_a_f32(arow, kb, grp), frag_b_bf16(bcol, kb), acc);
  }
  {
    const __bf16* arow = h_in + (size_t)(mBase + col) * NHsz;
    const __bf16* bcol = Wrec + (size_t)(nBase + col) * NHsz + grp * 16;
#pragma unroll 4
    for (int kb = 0; kb < NHsz; kb += 32)
      acc = wmma_bf16f32(frag_a_bf16(arow, kb, grp), frag_b_bf16(bcol, kb), acc);
  }

  const size_t base = (size_t)(mBase + grp * 8) * NHsz + nBase + col;
#pragma unroll
  for (int r = 0; r < 8; ++r) {
    const size_t idx = base + (size_t)r * NHsz;
    const float hn = (1.0f - s) * h_f32[idx] + s * av * tanhf(acc[r]);
    h_f32[idx] = hn;
    h_out[idx] = (__bf16)hn;
  }
}

__global__ __launch_bounds__(256) void prnn_decode_step_kernel(
    const __bf16* __restrict__ h,
    const __bf16* __restrict__ Wdec,
    const float* __restrict__ b_dec,
    float* __restrict__ y_t) {
  const int lane = threadIdx.x & 31;
  const int col  = lane & 15;
  const int grp  = lane >> 4;
  const int gw   = blockIdx.x * (blockDim.x >> 5) + (threadIdx.x >> 5);
  const int nTiles = NOsz / 16;
  const int mBase = (gw / nTiles) * 16;
  const int nBase = (gw % nTiles) * 16;

  const __bf16* arow = h + (size_t)(mBase + col) * NHsz;
  const __bf16* bcol = Wdec + (size_t)(nBase + col) * NHsz + grp * 16;

  v8f acc;
  const float bias = b_dec[nBase + col];
#pragma unroll
  for (int r = 0; r < 8; ++r) acc[r] = bias;

#pragma unroll 4
  for (int kb = 0; kb < NHsz; kb += 32)
    acc = wmma_bf16f32(frag_a_bf16(arow, kb, grp), frag_b_bf16(bcol, kb), acc);

  float* orow = y_t + (size_t)(mBase + grp * 8) * NOsz + nBase + col;
#pragma unroll
  for (int r = 0; r < 8; ++r) orow[(size_t)r * NOsz] = acc[r];
}

// ---------------- host launcher --------------------------------------------

extern "C" void kernel_launch(void* const* d_in, const int* in_sizes, int n_in,
                              void* d_out, int out_size, void* d_ws, size_t ws_size,
                              hipStream_t stream) {
  (void)in_sizes; (void)n_in; (void)out_size;
  const float* x     = (const float*)d_in[0];
  const float* dt    = (const float*)d_in[1];
  const float* a_g   = (const float*)d_in[2];
  const float* W_enc = (const float*)d_in[3];
  const float* b_enc = (const float*)d_in[4];
  const float* W_rec = (const float*)d_in[5];
  const float* b_rec = (const float*)d_in[6];
  const float* W_dec = (const float*)d_in[7];
  const float* b_dec = (const float*)d_in[8];

  float* y = (float*)d_out;                       // [T, B, NO]
  float* h_final_out = y + (size_t)TBsz * NOsz;   // [B, NH]

  char* ws = (char*)d_ws;
  size_t off = 0;
  auto take = [&](size_t bytes) -> char* {
    char* p = ws + off;
    off += (bytes + 255) & ~(size_t)255;
    return p;
  };
  __bf16* Wenc_h = (__bf16*)take((size_t)NHsz * NIsz * 2);
  __bf16* Wrec_h = (__bf16*)take((size_t)NHsz * NHsz * 2);
  __bf16* Wdec_h = (__bf16*)take((size_t)NOsz * NHsz * 2);
  float*  h_f32  = (float*) take((size_t)Bsz * NHsz * 4);
  __bf16* h_p0   = (__bf16*)take((size_t)Bsz * NHsz * 2);
  __bf16* h_p1   = (__bf16*)take((size_t)Bsz * NHsz * 2);
  __bf16* hist   = (__bf16*)take((size_t)TBsz * NHsz * 2);
  const size_t big_ws_needed = off;

  prnn_cvt_f32_to_bf16<<<1024, 256, 0, stream>>>(W_enc, Wenc_h, NHsz * NIsz);
  prnn_cvt_f32_to_bf16<<<2048, 256, 0, stream>>>(W_rec, Wrec_h, NHsz * NHsz);
  prnn_cvt_f32_to_bf16<<<512,  256, 0, stream>>>(W_dec, Wdec_h, NOsz * NHsz);

  prnn_zero_f32 <<<512, 256, 0, stream>>>(h_f32, Bsz * NHsz);
  prnn_zero_bf16<<<512, 256, 0, stream>>>(h_p0,  Bsz * NHsz);

  __bf16* ping[2] = {h_p0, h_p1};

  if (ws_size >= big_ws_needed) {
    // ---- fast path: hoisted encoder GEMM + scan + batched decoder ----
    prnn_encoder_kernel<<<(int)(TBsz / 16) * 8, 256, SMEM_I, stream>>>(
        x, Wenc_h, b_enc, hist);
    for (int t = 0; t < Tsz; ++t) {
      // 16 m-tiles x 8 n-groups = 128 blocks, 1024 tile-waves
      prnn_step_kernel<<<128, 256, SMEM_H, stream>>>(
          hist, ping[t & 1], ping[(t + 1) & 1], h_f32,
          Wrec_h, b_rec, dt, a_g, t);
    }
    prnn_decoder_kernel<<<(int)(TBsz / 16), 256, SMEM_H, stream>>>(
        hist, Wdec_h, b_dec, y);
  } else {
    // ---- fallback: fully fused per-step path (~16 MiB workspace) ----
    for (int t = 0; t < Tsz; ++t) {
      prnn_step_fused_kernel<<<256, 256, 0, stream>>>(
          x, Wenc_h, ping[t & 1], ping[(t + 1) & 1], h_f32,
          Wrec_h, b_enc, b_rec, dt, a_g, t);
      prnn_decode_step_kernel<<<32, 256, 0, stream>>>(
          ping[(t + 1) & 1], Wdec_h, b_dec, y + (size_t)t * Bsz * NOsz);
    }
  }

  hipMemcpyAsync(h_final_out, h_f32, (size_t)Bsz * NHsz * sizeof(float),
                 hipMemcpyDeviceToDevice, stream);
}
